// Text2SemanticDecoder_25237227831517
// MI455X (gfx1250) — compile-verified, matrix operand order checked by reading, not run on hardware
//
#include <hip/hip_runtime.h>

// ---------------------------------------------------------------------------
// Types / WMMA helpers (CDNA5 wave32, v_wmma_f32_16x16x32_bf16)
// ---------------------------------------------------------------------------
typedef __attribute__((ext_vector_type(16))) __bf16 v16bf;
typedef __attribute__((ext_vector_type(8)))  float  v8f;
typedef __attribute__((ext_vector_type(4)))  unsigned int uint32x4;
typedef __attribute__((ext_vector_type(8)))  int int32x8;
typedef __attribute__((ext_vector_type(4)))  int int32x4;

union FragU { unsigned int u[8]; v16bf v; };

__device__ __forceinline__ v8f wmma_bf16(v16bf a, v16bf b, v8f c) {
  return __builtin_amdgcn_wmma_f32_16x16x32_bf16(false, a, false, b, (short)0, c,
                                                 false, false);
}

__device__ __forceinline__ v8f v8f_zero() {
  v8f z;
#pragma unroll
  for (int i = 0; i < 8; i++) z[i] = 0.0f;
  return z;
}

// 16-bit A/B fragment interleave (ISA 7.12.2): uint j of the 8-uint fragment
// holds K = 2j,2j+1 with K(2j) = (2j<8 ? 2j : 2j+8) + half*8
__device__ __forceinline__ int uj_idx(int j, int half) {
  return (j < 4 ? j : j + 4) + half * 4;
}
// per-element K index for 16-element fragments
__device__ __forceinline__ int ke_idx(int e, int half) {
  return (e < 8 ? e : e + 8) + half * 8;
}

// ---------------------------------------------------------------------------
// TDM: async 2D bf16 tile load Global -> LDS with per-row padding.
// LDS layout produced: each row = 32 bf16 (16 DWORDs) then 1 pad DWORD
// => 17-uint row stride, matching the WMMA fragment reads.
// D# encoding per CDNA5 ISA (group0/group1; groups 2,3 = 0).
// This toolchain's builtin takes 6 args (g0, g1, g2, g3, g4, cpol).
// ---------------------------------------------------------------------------
__device__ __forceinline__ void tdm_load_tile_2d(unsigned lds_byte_off,
                                                 const __bf16* gptr,
                                                 unsigned tile_h,
                                                 unsigned long long stride_elems) {
  unsigned long long ga = (unsigned long long)(uintptr_t)gptr;
  uint32x4 g0;
  g0[0] = 1u;                                   // count=1, user descriptor
  g0[1] = lds_byte_off;                         // lds_addr (bytes)
  g0[2] = (unsigned)ga;                         // global_addr[31:0]
  g0[3] = (unsigned)((ga >> 32) & 0x01ffffffull) | (2u << 30);  // addr[56:32], type=2
  int32x8 g1;
  // workgroup_mask=0 | data_size=1(2B)<<16 | pad_enable<<20 |
  // pad_interval=3(16 DWORDs)<<22 | pad_amount=0(1 DWORD)<<25
  g1[0] = (int)0x00D10000u;
  // tensor_dim0 = tile_w (32): bits[79:48]; atomic_barrier_addr[15:0]=0
  g1[1] = (int)((32u & 0xffffu) << 16);
  // tensor_dim0 hi16 | tensor_dim1 lo16 (tensor_dim1 = tile_h, no OOB)
  g1[2] = (int)((tile_h & 0xffffu) << 16);
  // tensor_dim1 hi16 | tile_dim0 = 32
  g1[3] = (int)(32u << 16);
  // tile_dim1 = tile_h | tile_dim2 = 0
  g1[4] = (int)(tile_h & 0xffffu);
  // tensor_dim0_stride (48-bit, data_size units)
  g1[5] = (int)(unsigned)(stride_elems & 0xffffffffull);
  g1[6] = (int)(unsigned)((stride_elems >> 32) & 0xffffull);
  g1[7] = 0;
  int32x4 gz4 = {0, 0, 0, 0};
  int32x8 gz8 = {0, 0, 0, 0, 0, 0, 0, 0};
  __builtin_amdgcn_tensor_load_to_lds(g0, g1, gz4, gz4, gz8, 0);
}

// ---------------------------------------------------------------------------
// Problem constants
// ---------------------------------------------------------------------------
#define BB    8
#define SS    1024
#define DD    512
#define HH    8
#define HD    64
#define FF_   2048
#define STEPS 64
#define CAP   1088           // S + STEPS
#define NEGBIG (-1e30f)

// ---------------------------------------------------------------------------
// f32 -> bf16 conversion (grid-stride)
// ---------------------------------------------------------------------------
__global__ void cvt_f32_bf16(const float* __restrict__ in, __bf16* __restrict__ out,
                             long n) {
  long i = (long)blockIdx.x * blockDim.x + threadIdx.x;
  long stride = (long)gridDim.x * blockDim.x;
  for (; i < n; i += stride) out[i] = (__bf16)in[i];
}

// ---------------------------------------------------------------------------
// Tiled bf16 WMMA GEMM:  C[M,N] = act(A[M,K] @ W[N,K]^T + bias) (+ resid)
// Block: 256 threads (8 waves), block tile 128x64, K chunk 32.
// Tiles are DMA'd into LDS by the Tensor Data Mover, double-buffered:
// wave 0 issues next chunk's descriptors, waits TENSORcnt<=2 (in-order per
// wave => current chunk landed), then a block barrier publishes the LDS data.
// ---------------------------------------------------------------------------
template <bool RELU, bool RESID, bool OUT_BF16>
__global__ __launch_bounds__(256) void gemm_bf16(
    const __bf16* __restrict__ A, const __bf16* __restrict__ W,
    const float* __restrict__ bias, const float* __restrict__ resid,
    float* __restrict__ Cf, __bf16* __restrict__ Cb, int M, int N, int K) {
  __shared__ unsigned int As[2][128 * 17];  // 128 rows x 32 bf16 (+1 pad dword)
  __shared__ unsigned int Ws[2][64 * 17];   // 64 rows  x 32 bf16 (+1 pad dword)

  const int tid   = threadIdx.x;
  const int lane  = tid & 31;
  const int wave  = tid >> 5;            // 0..7
  const int waveM = wave >> 1;           // 0..3
  const int waveN = wave & 1;            // 0..1
  const int half  = (lane >> 4) & 1;
  const int l16   = lane & 15;

  const size_t blockM = (size_t)blockIdx.y * 128;
  const int    blockN = blockIdx.x * 64;

  const __bf16* Abase = A + blockM * (size_t)K;
  const __bf16* Wbase = W + (size_t)blockN * (size_t)K;

  v8f acc[2][2];
#pragma unroll
  for (int i = 0; i < 2; i++)
#pragma unroll
    for (int j = 0; j < 2; j++) acc[i][j] = v8f_zero();

  const int nch = K >> 5;

  // prologue: chunk 0 into buffer 0
  if (wave == 0) {
    tdm_load_tile_2d((unsigned)(uintptr_t)&As[0][0], Abase, 128, (unsigned)K);
    tdm_load_tile_2d((unsigned)(uintptr_t)&Ws[0][0], Wbase, 64, (unsigned)K);
  }

  for (int ch = 0; ch < nch; ch++) {
    const int cur = ch & 1;
    __syncthreads();  // previous compute on buffer cur^1 finished
    if (wave == 0) {
      if (ch + 1 < nch) {
        const int k1 = (ch + 1) << 5;
        tdm_load_tile_2d((unsigned)(uintptr_t)&As[cur ^ 1][0], Abase + k1, 128,
                         (unsigned)K);
        tdm_load_tile_2d((unsigned)(uintptr_t)&Ws[cur ^ 1][0], Wbase + k1, 64,
                         (unsigned)K);
        __builtin_amdgcn_s_wait_tensorcnt(2);  // chunk ch complete
      } else {
        __builtin_amdgcn_s_wait_tensorcnt(0);
      }
    }
    __syncthreads();  // publish buffer cur to all waves

    v16bf af[2], wf[2];
#pragma unroll
    for (int tm = 0; tm < 2; tm++) {
      FragU f;
      int ar = waveM * 32 + tm * 16 + l16;
#pragma unroll
      for (int j = 0; j < 8; j++) f.u[j] = As[cur][ar * 17 + uj_idx(j, half)];
      af[tm] = f.v;
    }
#pragma unroll
    for (int tn = 0; tn < 2; tn++) {
      FragU f;
      int wr = waveN * 32 + tn * 16 + l16;
#pragma unroll
      for (int j = 0; j < 8; j++) f.u[j] = Ws[cur][wr * 17 + uj_idx(j, half)];
      wf[tn] = f.v;
    }
#pragma unroll
    for (int tm = 0; tm < 2; tm++)
#pragma unroll
      for (int tn = 0; tn < 2; tn++)
        acc[tm][tn] = wmma_bf16(af[tm], wf[tn], acc[tm][tn]);
  }

  // epilogue: C layout — lane(0..15): M=r, lane(16..31): M=r+8, N = l16
#pragma unroll
  for (int tm = 0; tm < 2; tm++) {
#pragma unroll
    for (int tn = 0; tn < 2; tn++) {
      int col = blockN + waveN * 32 + tn * 16 + l16;
      float bcol = bias[col];
#pragma unroll
      for (int r = 0; r < 8; r++) {
        size_t row = blockM + waveM * 32 + tm * 16 + r + half * 8;
        float y = acc[tm][tn][r] + bcol;
        if (RELU) y = fmaxf(y, 0.0f);
        if (RESID) y += resid[row * (size_t)N + col];
        if (OUT_BF16)
          Cb[row * (size_t)N + col] = (__bf16)y;
        else
          Cf[row * (size_t)N + col] = y;
      }
    }
  }
}

// ---------------------------------------------------------------------------
// Split prefill qkv [B,S,3D] f32 -> q_bf [B,H,S,64], k/v caches [B,H,CAP,64] bf16
// ---------------------------------------------------------------------------
__global__ void extract_qkv(const float* __restrict__ qkv, __bf16* __restrict__ qbf,
                            __bf16* __restrict__ kc, __bf16* __restrict__ vc) {
  long i = (long)blockIdx.x * 256 + threadIdx.x;  // over B*S*D
  if (i >= (long)BB * SS * DD) return;
  long bs = i >> 9;           // b*S + s
  int d = (int)(i & 511);     // h*64 + hd
  long b = bs >> 10;
  int s = (int)(bs & 1023);
  int h = d >> 6, hd = d & 63;
  const float* src = qkv + bs * (3 * DD);
  long bh = b * HH + h;
  qbf[(bh * SS + s) * HD + hd]  = (__bf16)src[d];
  kc[(bh * CAP + s) * HD + hd]  = (__bf16)src[DD + d];
  vc[(bh * CAP + s) * HD + hd]  = (__bf16)src[2 * DD + d];
}

// ---------------------------------------------------------------------------
// Prefill causal attention (flash-style), one wave per (b,h, 16-query block).
// Scores and PV via v_wmma_f32_16x16x32_bf16, online softmax in f32.
// Writes attn output as bf16 in [B*S, D] layout.
// ---------------------------------------------------------------------------
__global__ __launch_bounds__(32) void prefill_attn(const __bf16* __restrict__ qbf,
                                                   const __bf16* __restrict__ kc,
                                                   const __bf16* __restrict__ vc,
                                                   __bf16* __restrict__ attn_bf) {
  const int bh = blockIdx.y;        // b*H + h
  const int b = bh >> 3, h = bh & 7;
  const int qb = blockIdx.x;        // 16-query block
  const int lane = threadIdx.x;
  const int half = lane >> 4, l16 = lane & 15;

  const unsigned int* Q  = (const unsigned int*)(qbf + (size_t)bh * SS * HD);
  const unsigned int* Kc = (const unsigned int*)(kc + (size_t)bh * CAP * HD);
  const __bf16* Vc = vc + (size_t)bh * CAP * HD;

  __shared__ unsigned int Pl[16 * 17];  // 16 x 34 bf16 transpose staging

  // Q fragments (dims 0..31 / 32..63), row = qb*16 + l16, 32 uints per row
  v16bf qf[2];
  {
    int qrow = qb * 16 + l16;
#pragma unroll
    for (int c = 0; c < 2; c++) {
      FragU f;
#pragma unroll
      for (int j = 0; j < 8; j++) f.u[j] = Q[qrow * 32 + uj_idx(j, half) + c * 16];
      qf[c] = f.v;
    }
  }

  v8f accO[4];
#pragma unroll
  for (int n = 0; n < 4; n++) accO[n] = v8f_zero();
  float rmax[8], rsum[8];
#pragma unroll
  for (int r = 0; r < 8; r++) { rmax[r] = NEGBIG; rsum[r] = 0.0f; }

  const float scale = 0.125f;  // 1/sqrt(64)
  const int nchunks = (qb * 16 + 15) / 32 + 1;

  for (int ch = 0; ch < nchunks; ch++) {
    const int base = ch * 32;

    // K fragments for two 16-key tiles
    v16bf kf0[2], kf1[2];
#pragma unroll
    for (int c = 0; c < 2; c++) {
      FragU f0, f1;
      int r0 = base + l16, r1 = base + 16 + l16;
#pragma unroll
      for (int j = 0; j < 8; j++) {
        f0.u[j] = Kc[r0 * 32 + uj_idx(j, half) + c * 16];
        f1.u[j] = Kc[r1 * 32 + uj_idx(j, half) + c * 16];
      }
      kf0[c] = f0.v;
      kf1[c] = f1.v;
    }

    v8f s0 = wmma_bf16(qf[0], kf0[0], v8f_zero());
    s0 = wmma_bf16(qf[1], kf0[1], s0);
    v8f s1 = wmma_bf16(qf[0], kf1[0], v8f_zero());
    s1 = wmma_bf16(qf[1], kf1[1], s1);

    float pv0[8], pv1[8];
#pragma unroll
    for (int r = 0; r < 8; r++) {
      int qi = qb * 16 + r + half * 8;
      int k0i = base + l16, k1i = base + 16 + l16;
      float v0 = s0[r] * scale; if (k0i > qi) v0 = NEGBIG;
      float v1 = s1[r] * scale; if (k1i > qi) v1 = NEGBIG;
      // row max across the 16 lanes of this half
      float m = fmaxf(v0, v1);
#pragma unroll
      for (int off = 8; off >= 1; off >>= 1) m = fmaxf(m, __shfl_xor(m, off, 32));
      m = fmaxf(m, rmax[r]);
      float corr = __expf(rmax[r] - m);
      rmax[r] = m;
      float e0 = __expf(v0 - m);
      float e1 = __expf(v1 - m);
      float sm = e0 + e1;
#pragma unroll
      for (int off = 8; off >= 1; off >>= 1) sm += __shfl_xor(sm, off, 32);
      rsum[r] = rsum[r] * corr + sm;
      pv0[r] = e0; pv1[r] = e1;
#pragma unroll
      for (int n = 0; n < 4; n++) accO[n][r] *= corr;
    }

    // C-layout -> A-layout transpose of P through LDS (single wave)
    __syncthreads();
    {
      __bf16* Pb = (__bf16*)Pl;
#pragma unroll
      for (int r = 0; r < 8; r++) {
        int M = r + half * 8;
        Pb[M * 34 + l16]      = (__bf16)pv0[r];
        Pb[M * 34 + 16 + l16] = (__bf16)pv1[r];
      }
    }
    __syncthreads();
    FragU pf;
#pragma unroll
    for (int j = 0; j < 8; j++) pf.u[j] = Pl[l16 * 17 + uj_idx(j, half)];

    // V B-fragments (K = key, N = head dim group of 16)
#pragma unroll
    for (int n = 0; n < 4; n++) {
      FragU vf;
#pragma unroll
      for (int e = 0; e < 16; e++) {
        int kk = ke_idx(e, half);
        vf.v[e] = Vc[(size_t)(base + kk) * HD + n * 16 + l16];
      }
      accO[n] = wmma_bf16(pf.v, vf.v, accO[n]);
    }
  }

  // normalize and write bf16 output [B*S, D]
#pragma unroll
  for (int r = 0; r < 8; r++) {
    float inv = 1.0f / rsum[r];
    size_t row = (size_t)b * SS + qb * 16 + r + half * 8;
#pragma unroll
    for (int n = 0; n < 4; n++) {
      attn_bf[row * DD + h * HD + n * 16 + l16] = (__bf16)(accO[n][r] * inv);
    }
  }
}

// ---------------------------------------------------------------------------
// LayerNorm over D=512, one block per row. Optional bf16 mirror output.
// out_row = (row / rpb) * obs + row % rpb  (row remap for writing into d_out)
// ---------------------------------------------------------------------------
__global__ void layernorm512(const float* __restrict__ in, const float* __restrict__ g,
                             const float* __restrict__ be, float* __restrict__ out32,
                             __bf16* __restrict__ outbf, int rpb, long obs) {
  const int row = blockIdx.x;
  const int tid = threadIdx.x;  // 256
  const float* x = in + (size_t)row * DD;
  __shared__ float red[256];

  float v0 = x[tid], v1 = x[tid + 256];
  red[tid] = v0 + v1;
  __syncthreads();
  for (int s = 128; s > 0; s >>= 1) {
    if (tid < s) red[tid] += red[tid + s];
    __syncthreads();
  }
  float mean = red[0] * (1.0f / DD);
  __syncthreads();
  float d0 = v0 - mean, d1 = v1 - mean;
  red[tid] = d0 * d0 + d1 * d1;
  __syncthreads();
  for (int s = 128; s > 0; s >>= 1) {
    if (tid < s) red[tid] += red[tid + s];
    __syncthreads();
  }
  float inv = rsqrtf(red[0] * (1.0f / DD) + 1e-5f);

  long orow = (long)(row / rpb) * obs + (row % rpb);
  float y0 = d0 * inv * g[tid] + be[tid];
  float y1 = d1 * inv * g[tid + 256] + be[tid + 256];
  if (out32) {
    out32[orow * DD + tid] = y0;
    out32[orow * DD + tid + 256] = y1;
  }
  if (outbf) {
    outbf[orow * DD + tid] = (__bf16)y0;
    outbf[orow * DD + tid + 256] = (__bf16)y1;
  }
}

// ---------------------------------------------------------------------------
// Decode GEMV: one wave per output element. Y[r,n] = act(A[r,:]·W[n,:]+bias)+resid
// ---------------------------------------------------------------------------
template <bool RELU, bool RESID>
__global__ void gemv_bf16w(const float* __restrict__ A, int lda,
                           const __bf16* __restrict__ W, const float* __restrict__ bias,
                           const float* __restrict__ resid, float* __restrict__ Y,
                           int rows, int N, int K) {
  int wid = blockIdx.x * 8 + (threadIdx.x >> 5);
  int lane = threadIdx.x & 31;
  if (wid >= rows * N) return;
  int r = wid / N, n = wid % N;
  const float* a = A + (size_t)r * lda;
  const __bf16* w = W + (size_t)n * K;
  float s = 0.0f;
  for (int k = lane; k < K; k += 32) s += a[k] * (float)w[k];
#pragma unroll
  for (int off = 16; off >= 1; off >>= 1) s += __shfl_xor(s, off, 32);
  if (lane == 0) {
    float y = s + bias[n];
    if (RELU) y = fmaxf(y, 0.0f);
    if (RESID) y += resid[(size_t)r * N + n];
    Y[(size_t)r * N + n] = y;
  }
}

// ---------------------------------------------------------------------------
// Decode: append new k/v (with bias already applied) into bf16 caches at `pos`
// ---------------------------------------------------------------------------
__global__ void decode_append(const float* __restrict__ qkv_dec,
                              __bf16* __restrict__ kc, __bf16* __restrict__ vc,
                              int pos) {
  int i = blockIdx.x * 256 + threadIdx.x;  // B*D
  if (i >= BB * DD) return;
  int b = i >> 9, d = i & 511;
  int h = d >> 6, hd = d & 63;
  long bh = (long)b * HH + h;
  kc[(bh * CAP + pos) * HD + hd] = (__bf16)qkv_dec[(size_t)b * 3 * DD + DD + d];
  vc[(bh * CAP + pos) * HD + hd] = (__bf16)qkv_dec[(size_t)b * 3 * DD + 2 * DD + d];
}

// ---------------------------------------------------------------------------
// Decode attention: one block per (b,h), L keys. q read f32 from qkv_dec.
// ---------------------------------------------------------------------------
__global__ void decode_attn(const float* __restrict__ qkv_dec,
                            const __bf16* __restrict__ kc,
                            const __bf16* __restrict__ vc,
                            float* __restrict__ attn_dec, int L) {
  const int bh = blockIdx.x;
  const int b = bh >> 3, h = bh & 7;
  const float* q = qkv_dec + (size_t)b * 3 * DD + h * HD;
  const __bf16* K = kc + (size_t)bh * CAP * HD;
  const __bf16* V = vc + (size_t)bh * CAP * HD;
  const int tid = threadIdx.x;  // 256

  __shared__ float sc[CAP];
  __shared__ float red[256];

  float lmax = NEGBIG;
  for (int j = tid; j < L; j += 256) {
    float s = 0.0f;
#pragma unroll 8
    for (int d = 0; d < HD; d++) s += q[d] * (float)K[(size_t)j * HD + d];
    s *= 0.125f;
    sc[j] = s;
    lmax = fmaxf(lmax, s);
  }
  red[tid] = lmax;
  __syncthreads();
  for (int s = 128; s > 0; s >>= 1) {
    if (tid < s) red[tid] = fmaxf(red[tid], red[tid + s]);
    __syncthreads();
  }
  float m = red[0];
  __syncthreads();

  float lsum = 0.0f;
  for (int j = tid; j < L; j += 256) {
    float e = __expf(sc[j] - m);
    sc[j] = e;
    lsum += e;
  }
  red[tid] = lsum;
  __syncthreads();
  for (int s = 128; s > 0; s >>= 1) {
    if (tid < s) red[tid] += red[tid + s];
    __syncthreads();
  }
  float inv = 1.0f / red[0];
  __syncthreads();

  const int d = tid & 63, slice = tid >> 6;  // 64 dims x 4 slices
  float acc = 0.0f;
  for (int j = slice; j < L; j += 4) acc += sc[j] * (float)V[(size_t)j * HD + d];
  red[tid] = acc;
  __syncthreads();
  if (slice == 0) {
    float a = red[d] + red[64 + d] + red[128 + d] + red[192 + d];
    attn_dec[(size_t)b * DD + h * HD + d] = a * inv;
  }
}

// ---------------------------------------------------------------------------
// Host orchestration
// ---------------------------------------------------------------------------
extern "C" void kernel_launch(void* const* d_in, const int* in_sizes, int n_in,
                              void* d_out, int out_size, void* d_ws, size_t ws_size,
                              hipStream_t stream) {
  (void)in_sizes; (void)n_in; (void)out_size; (void)ws_size;
  const float* x        = (const float*)d_in[0];   // [B,S,D]
  const float* decode_x = (const float*)d_in[1];   // [STEPS,B,1,D]
  // d_in[2] causal_mask: handled analytically
  const float* qkv_w = (const float*)d_in[3];
  const float* qkv_b = (const float*)d_in[4];
  const float* out_w = (const float*)d_in[5];
  const float* out_b = (const float*)d_in[6];
  const float* w1    = (const float*)d_in[7];
  const float* b1    = (const float*)d_in[8];
  const float* w2    = (const float*)d_in[9];
  const float* b2    = (const float*)d_in[10];
  const float* ln1w  = (const float*)d_in[11];
  const float* ln1b  = (const float*)d_in[12];
  const float* ln2w  = (const float*)d_in[13];
  const float* ln2b  = (const float*)d_in[14];
  float* out = (float*)d_out;                      // [B, S+STEPS, D]

  char* ws = (char*)d_ws;
  // ---- workspace arena (bytes) ----
  const size_t O_WQKV = 0;                              // 1536*512 bf16
  const size_t O_WOUT = O_WQKV + 1536 * 512 * 2;        // 512*512 bf16
  const size_t O_W1   = O_WOUT + 512 * 512 * 2;         // 2048*512 bf16
  const size_t O_W2   = O_W1 + 2048 * 512 * 2;          // 512*2048 bf16
  const size_t O_XBF  = O_W2 + 512 * 2048 * 2;          // 8192*512 bf16 (alias attn_bf)
  const size_t O_QKV  = O_XBF + (size_t)8192 * 512 * 2; // 8192*1536 f32 arena
  const size_t O_FFBF = O_QKV;                          // 8192*2048 bf16 (aliases qkv)
  const size_t O_H2   = O_QKV + (size_t)8192 * 2048 * 2;// 8192*512 f32 (aliases qkv)
  const size_t O_QBF  = O_QKV + (size_t)8192 * 1536 * 4;// 8*8*1024*64 bf16
  const size_t O_KC   = O_QBF + (size_t)BB * HH * SS * HD * 2;  // [B,H,CAP,64] bf16
  const size_t O_VC   = O_KC + (size_t)BB * HH * CAP * HD * 2;
  const size_t O_H1   = O_VC + (size_t)BB * HH * CAP * HD * 2;  // 8192*512 f32
  const size_t O_H1BF = O_H1 + (size_t)8192 * 512 * 4;          // 8192*512 bf16
  const size_t O_DEC  = O_H1BF + (size_t)8192 * 512 * 2;        // decode scratch

  __bf16* wqkv_bf = (__bf16*)(ws + O_WQKV);
  __bf16* wout_bf = (__bf16*)(ws + O_WOUT);
  __bf16* w1_bf   = (__bf16*)(ws + O_W1);
  __bf16* w2_bf   = (__bf16*)(ws + O_W2);
  __bf16* x_bf    = (__bf16*)(ws + O_XBF);
  __bf16* attn_bf = (__bf16*)(ws + O_XBF);   // alias (x_bf dead after qkv GEMM)
  float*  qkv32   = (float*)(ws + O_QKV);
  __bf16* ff_bf   = (__bf16*)(ws + O_FFBF);  // alias (qkv32 dead after extract)
  float*  h2      = (float*)(ws + O_H2);     // alias
  __bf16* q_bf    = (__bf16*)(ws + O_QBF);
  __bf16* kcache  = (__bf16*)(ws + O_KC);
  __bf16* vcache  = (__bf16*)(ws + O_VC);
  float*  h1      = (float*)(ws + O_H1);
  __bf16* h1_bf   = (__bf16*)(ws + O_H1BF);
  float*  qkv_dec = (float*)(ws + O_DEC);                 // 8*1536
  float*  attn_dec= (float*)(ws + O_DEC + 49152);         // 8*512
  float*  h_dec   = (float*)(ws + O_DEC + 65536);         // 8*512
  float*  ff_dec  = (float*)(ws + O_DEC + 81920);         // 8*2048
  float*  h2_dec  = (float*)(ws + O_DEC + 147456);        // 8*512

  auto cvt = [&](const float* src, __bf16* dst, long n) {
    int grid = (int)((n + 255) / 256);
    if (grid > 4096) grid = 4096;
    cvt_f32_bf16<<<grid, 256, 0, stream>>>(src, dst, n);
  };

  // ---- weight + activation conversion ----
  cvt(qkv_w, wqkv_bf, (long)1536 * 512);
  cvt(out_w, wout_bf, (long)512 * 512);
  cvt(w1, w1_bf, (long)2048 * 512);
  cvt(w2, w2_bf, (long)512 * 2048);
  cvt(x, x_bf, (long)8192 * 512);

  // ---- prefill ----
  gemm_bf16<false, false, false><<<dim3(1536 / 64, 8192 / 128), 256, 0, stream>>>(
      x_bf, wqkv_bf, qkv_b, nullptr, qkv32, nullptr, 8192, 1536, 512);

  extract_qkv<<<(int)(((long)BB * SS * DD + 255) / 256), 256, 0, stream>>>(
      qkv32, q_bf, kcache, vcache);

  prefill_attn<<<dim3(SS / 16, BB * HH), 32, 0, stream>>>(q_bf, kcache, vcache,
                                                          attn_bf);

  gemm_bf16<false, true, false><<<dim3(512 / 64, 8192 / 128), 256, 0, stream>>>(
      attn_bf, wout_bf, out_b, x, h1, nullptr, 8192, 512, 512);

  layernorm512<<<8192, 256, 0, stream>>>(h1, ln1w, ln1b, h1, h1_bf, 8192, 0);

  gemm_bf16<true, false, true><<<dim3(2048 / 64, 8192 / 128), 256, 0, stream>>>(
      h1_bf, w1_bf, b1, nullptr, nullptr, ff_bf, 8192, 2048, 512);

  gemm_bf16<false, true, false><<<dim3(512 / 64, 8192 / 128), 256, 0, stream>>>(
      ff_bf, w2_bf, b2, h1, h2, nullptr, 8192, 512, 2048);

  // LN2 -> d_out rows (b*1088 + s)
  layernorm512<<<8192, 256, 0, stream>>>(h2, ln2w, ln2b, out, nullptr, 1024, 1088);

  // ---- autoregressive decode ----
  for (int t = 0; t < STEPS; t++) {
    const float* xt = decode_x + (size_t)t * BB * DD;
    const int pos = SS + t;

    gemv_bf16w<false, false><<<(BB * 1536) / 8, 256, 0, stream>>>(
        xt, DD, wqkv_bf, qkv_b, nullptr, qkv_dec, BB, 1536, 512);

    decode_append<<<(BB * DD + 255) / 256, 256, 0, stream>>>(qkv_dec, kcache,
                                                             vcache, pos);

    decode_attn<<<BB * HH, 256, 0, stream>>>(qkv_dec, kcache, vcache, attn_dec,
                                             pos + 1);

    gemv_bf16w<false, true><<<(BB * 512) / 8, 256, 0, stream>>>(
        attn_dec, DD, wout_bf, out_b, xt, h_dec, BB, 512, 512);

    layernorm512<<<BB, 256, 0, stream>>>(h_dec, ln1w, ln1b, h_dec, nullptr, 8, 0);

    gemv_bf16w<true, false><<<(BB * 2048) / 8, 256, 0, stream>>>(
        h_dec, DD, w1_bf, b1, nullptr, ff_dec, BB, 2048, 512);

    gemv_bf16w<false, true><<<(BB * 512) / 8, 256, 0, stream>>>(
        ff_dec, 2048, w2_bf, b2, h_dec, h2_dec, BB, 512, 2048);

    layernorm512<<<BB, 256, 0, stream>>>(h2_dec, ln2w, ln2b,
                                         out + (size_t)pos * DD, nullptr, 1, 1088);
  }
}
